// GNNStack_55817394979044
// MI455X (gfx1250) — compile-verified, matrix operand order checked by reading, not run on hardware
//
#include <hip/hip_runtime.h>
#include <math.h>

// ---------------- problem constants (match reference) ----------------
constexpr int N   = 50000;   // nodes
constexpr int E   = 800000;  // edges
constexpr int NE  = 200000;  // eval edges
constexpr int DIN = 128;     // input dim
constexpr int H   = 4;       // heads
constexpr int C   = 32;      // per-head channels
constexpr int HC  = H * C;   // 128
constexpr float SLOPE = 0.2f;

// ---------------- WMMA vector types ----------------
typedef __attribute__((ext_vector_type(16))) __bf16 v16bf;
typedef __attribute__((ext_vector_type(8)))  __bf16 v8bf;
typedef __attribute__((ext_vector_type(8)))  float  v8f;

// ---------------- helpers ----------------
__device__ __forceinline__ unsigned enc_f32(float f) {
  unsigned u = __float_as_uint(f);
  return (u >> 31) ? ~u : (u | 0x80000000u);   // order-preserving encode
}
__device__ __forceinline__ float dec_f32(unsigned k) {
  unsigned u = (k >> 31) ? (k ^ 0x80000000u) : ~k;
  return __uint_as_float(u);
}

// ---------------- 1) f32 -> bf16 copy of node features ----------------
__global__ void k_cvt_x(const float* __restrict__ src, __bf16* __restrict__ dst, int n) {
  int i = blockIdx.x * blockDim.x + threadIdx.x;
  if (i < n) dst[i] = (__bf16)src[i];
}

// ---------------- 2) transpose + convert weight: wbT[c*128+k] = w[k*128+c] ----
__global__ void k_cvt_w(const float* __restrict__ w, __bf16* __restrict__ wbT) {
  int i = blockIdx.x * blockDim.x + threadIdx.x;   // i < 128*128
  int c = i >> 7, k = i & 127;
  wbT[c * DIN + k] = (__bf16)w[k * HC + c];
}

// ---------------- 3) GEMM: h = A(bf16, M x 128) * W(bf16) + lin_b -----------
// Block = 8 waves; each wave computes one 16x16 tile (v_wmma_f32_16x16x32_bf16).
// The shared 16x128 A tile (4 KB, contiguous) is staged into LDS once per block
// with GLOBAL_LOAD_ASYNC_TO_LDS_B128 (ASYNCcnt), then read back via ds_load.
__global__ __launch_bounds__(256) void k_gemm(const __bf16* __restrict__ A,
                                              const __bf16* __restrict__ BT,
                                              const float* __restrict__ bias,
                                              float* __restrict__ Cm) {
  __shared__ __align__(16) __bf16 Atile[16 * DIN];   // 4 KB

  const int tid  = threadIdx.x;
  const int wave = tid >> 5;                  // 0..7 -> 16-col tile
  const int lane = tid & 31;
  const int rowBase = blockIdx.x * 16;        // 3125 blocks, exact
  const int colBase = wave * 16;
  const int lrow  = lane & 15;
  const int khalf = (lane < 16) ? 0 : 8;      // per-ISA 16-bit A layout

  // --- async copy of the block's A tile: 256 thr x 16 B = 4096 B ---
  {
    unsigned ldsoff = (unsigned)(size_t)(&Atile[0]) + (unsigned)tid * 16u;
    const __bf16* gsrc = A + (size_t)rowBase * DIN + (size_t)tid * 8;
    asm volatile(
        "global_load_async_to_lds_b128 %0, %1, off\n\t"
        "s_wait_asynccnt 0x0"
        :: "v"(ldsoff), "v"(gsrc) : "memory");
  }
  __syncthreads();

  const __bf16* arow = &Atile[lrow * DIN];                    // LDS
  const __bf16* bcol = BT + (size_t)(colBase + lrow) * DIN;   // global (L2-resident)

  v8f acc = {};
#pragma unroll
  for (int k0 = 0; k0 < DIN; k0 += 32) {
    v8bf a_lo = *(const v8bf*)(arow + k0 + khalf);
    v8bf a_hi = *(const v8bf*)(arow + k0 + khalf + 16);
    v8bf b_lo = *(const v8bf*)(bcol + k0 + khalf);
    v8bf b_hi = *(const v8bf*)(bcol + k0 + khalf + 16);
    v16bf a = __builtin_shufflevector(a_lo, a_hi, 0,1,2,3,4,5,6,7,8,9,10,11,12,13,14,15);
    v16bf b = __builtin_shufflevector(b_lo, b_hi, 0,1,2,3,4,5,6,7,8,9,10,11,12,13,14,15);
    acc = __builtin_amdgcn_wmma_f32_16x16x32_bf16(false, a, false, b,
                                                  (short)0, acc, false, false);
  }

  const int col  = colBase + lrow;
  const float bv = bias[col];
  const int rtop = rowBase + ((lane < 16) ? 0 : 8);
#pragma unroll
  for (int v = 0; v < 8; ++v)
    Cm[(size_t)(rtop + v) * HC + col] = acc[v] + bv;
}

// ---------------- 4) init segment buffers + aggregation target --------------
__global__ void k_init(float* __restrict__ aggr, unsigned* __restrict__ smax,
                       float* __restrict__ denom, int n_feat, int n_head) {
  int i = blockIdx.x * blockDim.x + threadIdx.x;
  if (i < n_feat) aggr[i] = 0.f;
  if (i < n_head) { smax[i] = 0u; denom[i] = 0.f; }
}

// ---------------- 5) per-edge attention scores + segment max ----------------
// warp per edge; lane = channel within head
__global__ __launch_bounds__(256) void k_scores(const float* __restrict__ h,
                                                const int* __restrict__ ei,
                                                const float* __restrict__ att,
                                                float* __restrict__ scores,
                                                unsigned* __restrict__ smax) {
  int t = blockIdx.x * blockDim.x + threadIdx.x;
  int e = t >> 5, lane = t & 31;
  if (e >= E) return;
  int src = ei[e], dst = ei[E + e];
  const float* hs = h + (size_t)src * HC;
  const float* hd = h + (size_t)dst * HC;
  float sc[H];
#pragma unroll
  for (int i = 0; i < H; ++i) {
    int f = i * 32 + lane;
    sc[i] = hs[f] * att[i * 64 + lane] + hd[f] * att[i * 64 + 32 + lane];
  }
#pragma unroll
  for (int o = 16; o; o >>= 1)
#pragma unroll
    for (int i = 0; i < H; ++i) sc[i] += __shfl_down(sc[i], o, 32);
  if (lane == 0) {
#pragma unroll
    for (int i = 0; i < H; ++i) {
      float s = sc[i] > 0.f ? sc[i] : SLOPE * sc[i];
      scores[(size_t)e * H + i] = s;
      atomicMax(&smax[dst * H + i], enc_f32(s));
    }
  }
}

// ---------------- 6) alpha = exp(score - max); denom += alpha ---------------
__global__ void k_alpha(const int* __restrict__ ei, float* __restrict__ scores,
                        const unsigned* __restrict__ smax,
                        float* __restrict__ denom) {
  int idx = blockIdx.x * blockDim.x + threadIdx.x;
  if (idx >= E * H) return;
  int e = idx >> 2, i = idx & 3;
  int dst = ei[E + e];
  float a = expf(scores[idx] - dec_f32(smax[dst * H + i]));
  scores[idx] = a;
  atomicAdd(&denom[dst * H + i], a);
}

// ---------------- 7) message aggregation: aggr[dst] += h[src]*alpha ---------
__global__ __launch_bounds__(256) void k_aggr(const float* __restrict__ h,
                                              const int* __restrict__ ei,
                                              const float* __restrict__ scores,
                                              const float* __restrict__ denom,
                                              float* __restrict__ aggr) {
  int t = blockIdx.x * blockDim.x + threadIdx.x;
  int e = t >> 5, lane = t & 31;
  if (e >= E) return;
  int src = ei[e], dst = ei[E + e];
  // single b128 broadcast loads instead of 4 scalar loads
  const float4 sv = *(const float4*)(scores + (size_t)e * H);
  const float4 dv = *(const float4*)(denom + (size_t)dst * H);
  float w[H] = { sv.x / (dv.x + 1e-16f), sv.y / (dv.y + 1e-16f),
                 sv.z / (dv.z + 1e-16f), sv.w / (dv.w + 1e-16f) };
#pragma unroll
  for (int i = 0; i < H; ++i) {
    int f = i * 32 + lane;
    atomicAdd(&aggr[(size_t)dst * HC + f], h[(size_t)src * HC + f] * w[i]);
  }
}

// ---------------- 8) x = relu(aggr + bias) ----------------------------------
__global__ void k_finalize(float* __restrict__ x, const float* __restrict__ bias, int n) {
  int i = blockIdx.x * blockDim.x + threadIdx.x;
  if (i >= n) return;
  float v = x[i] + bias[i & (HC - 1)];
  x[i] = v > 0.f ? v : 0.f;
}

// ---------------- 9) edge MLP head + log_softmax ----------------------------
__global__ __launch_bounds__(256) void k_head(const float* __restrict__ x,
                                              const int* __restrict__ ev,
                                              const float* __restrict__ w1,
                                              const float* __restrict__ b1,
                                              const float* __restrict__ w2,
                                              const float* __restrict__ b2,
                                              float* __restrict__ out) {
  __shared__ float ec[8][2 * HC];
  int warp = threadIdx.x >> 5, lane = threadIdx.x & 31;
  int e = blockIdx.x * 8 + warp;
  if (e >= NE) return;
  int a = ev[e], b = ev[NE + e];
#pragma unroll
  for (int i = 0; i < 4; ++i) ec[warp][i * 32 + lane]      = x[(size_t)a * HC + i * 32 + lane];
#pragma unroll
  for (int i = 0; i < 4; ++i) ec[warp][HC + i * 32 + lane] = x[(size_t)b * HC + i * 32 + lane];
  // lane owns one of 32 hidden cols
  float o1 = b1[lane];
  for (int k = 0; k < 2 * HC; ++k) o1 += ec[warp][k] * w1[k * C + lane];
  float p0 = o1 * w2[lane * 2 + 0];
  float p1 = o1 * w2[lane * 2 + 1];
#pragma unroll
  for (int o = 16; o; o >>= 1) { p0 += __shfl_down(p0, o, 32); p1 += __shfl_down(p1, o, 32); }
  if (lane == 0) {
    p0 += b2[0]; p1 += b2[1];
    float m = fmaxf(p0, p1);
    float lse = m + logf(expf(p0 - m) + expf(p1 - m));
    out[(size_t)e * 2 + 0] = p0 - lse;
    out[(size_t)e * 2 + 1] = p1 - lse;
  }
}

// ---------------- launcher ----------------
extern "C" void kernel_launch(void* const* d_in, const int* in_sizes, int n_in,
                              void* d_out, int out_size, void* d_ws, size_t ws_size,
                              hipStream_t stream) {
  const float* x_in   = (const float*)d_in[0];
  const int*   ei     = (const int*)d_in[1];
  const int*   ev     = (const int*)d_in[2];
  const float* lin_w  = (const float*)d_in[3];   // [2,128,128]
  const float* lin_b  = (const float*)d_in[4];   // [2,128]
  const float* att    = (const float*)d_in[5];   // [2,4,64]
  const float* bias   = (const float*)d_in[6];   // [2,128]
  const float* pw1    = (const float*)d_in[7];   // [256,32]
  const float* pb1    = (const float*)d_in[8];
  const float* pw2    = (const float*)d_in[9];   // [32,2]
  const float* pb2    = (const float*)d_in[10];
  float* out = (float*)d_out;

  // workspace carve (256B aligned)
  char* ws = (char*)d_ws;
  size_t off = 0;
  auto carve = [&](size_t bytes) { char* p = ws + off; off = (off + bytes + 255) & ~(size_t)255; return p; };
  __bf16*   xb    = (__bf16*)  carve((size_t)N * DIN * 2);
  __bf16*   wbT   = (__bf16*)  carve((size_t)HC * DIN * 2);
  float*    hbuf  = (float*)   carve((size_t)N * HC * 4);
  float*    scores= (float*)   carve((size_t)E * H * 4);
  unsigned* smax  = (unsigned*)carve((size_t)N * H * 4);
  float*    denom = (float*)   carve((size_t)N * H * 4);
  float*    x1    = (float*)   carve((size_t)N * HC * 4);
  float*    x2    = (float*)   carve((size_t)N * HC * 4);
  (void)ws_size; (void)in_sizes; (void)n_in; (void)out_size;

  const int nf = N * HC;            // 6.4M node-feature elems
  const int nh = N * H;
  const dim3 B256(256);

  const float* xcur = x_in;
  float* xnext = x1;
  for (int l = 0; l < 2; ++l) {
    // h = bf16(xcur) @ bf16(lin_w[l]) + lin_b[l]   (WMMA bf16)
    k_cvt_x<<<(nf + 255) / 256, B256, 0, stream>>>(xcur, xb, nf);
    k_cvt_w<<<(DIN * HC) / 256, B256, 0, stream>>>(lin_w + (size_t)l * DIN * HC, wbT);
    k_gemm<<<N / 16, B256, 0, stream>>>(xb, wbT, lin_b + (size_t)l * HC, hbuf);
    // attention softmax over incoming edges
    k_init<<<(nf + 255) / 256, B256, 0, stream>>>(xnext, smax, denom, nf, nh);
    k_scores<<<(E * 32) / 256, B256, 0, stream>>>(hbuf, ei, att + (size_t)l * H * 2 * C, scores, smax);
    k_alpha<<<(E * H) / 256, B256, 0, stream>>>(ei, scores, smax, denom);
    k_aggr<<<(E * 32) / 256, B256, 0, stream>>>(hbuf, ei, scores, denom, xnext);
    k_finalize<<<(nf + 255) / 256, B256, 0, stream>>>(xnext, bias + (size_t)l * HC, nf);
    xcur = xnext;
    xnext = x2;
  }

  // edge MLP head + log_softmax on eval edges
  k_head<<<(NE + 7) / 8, B256, 0, stream>>>(xcur, ev, pw1, pb1, pw2, pb2, out);
}
